// Model_2S_SGCN_10505490006519
// MI455X (gfx1250) — matrix operand order repaired and checked by reading, not compiled, not run on hardware
//
#include <hip/hip_runtime.h>
#include <hip/hip_bf16.h>

typedef __attribute__((ext_vector_type(16))) _Float16 v16h;
typedef __attribute__((ext_vector_type(8)))  _Float16 v8h;
typedef __attribute__((ext_vector_type(8)))  float    v8f;

// ---------------------------------------------------------------- utilities

__global__ void fill_kernel(float* __restrict__ p, float v, int n) {
    int i = blockIdx.x * blockDim.x + threadIdx.x;
    int s = gridDim.x * blockDim.x;
    for (; i < n; i += s) p[i] = v;
}

__global__ void deg_count_kernel(const int* __restrict__ dst, float* __restrict__ deg, int E) {
    int i = blockIdx.x * blockDim.x + threadIdx.x;
    int s = gridDim.x * blockDim.x;
    for (; i < E; i += s) atomicAdd(&deg[dst[i]], 1.0f);
}

__global__ void rsqrt_kernel(float* __restrict__ p, int n) {
    int i = blockIdx.x * blockDim.x + threadIdx.x;
    int s = gridDim.x * blockDim.x;
    for (; i < n; i += s) p[i] = rsqrtf(p[i]);
}

// ------------------------------------------------- weight pre-split (fp32 -> transposed f16 hi/lo)
// W is [K][Nc] row-major with row stride ldw; produce Th/Tl as [Nc][K] (K contiguous).

__global__ void convert_wt_kernel(const float* __restrict__ W, int ldw, int K, int Nc,
                                  _Float16* __restrict__ Th, _Float16* __restrict__ Tl)
{
    int i = blockIdx.x * blockDim.x + threadIdx.x;
    int s = gridDim.x * blockDim.x;
    int tot = K * Nc;
    for (; i < tot; i += s) {
        int n = i / K, k = i - n * K;
        float v = W[(size_t)k * ldw + n];
        _Float16 h = (_Float16)v;
        Th[i] = h;
        Tl[i] = (_Float16)(v - (float)h);
    }
}

// ------------------------------------------------- split-f16 WMMA GEMM (fp32 A/C, pre-split B^T)
// C[M x NT*16] = A[M x K] * B[K x NT*16]  (+ bias, + old C if accumulate)
// One wave computes a 16-row strip. B supplied transposed as f16 hi/lo: BT*[n][k], K contiguous,
// so each lane's B fragment is two 16-byte runs 32 bytes apart (2x b128). K is a compile-time constant so the
// k-loop fully unrolls and all fragment addresses become immediate offsets off one base pointer.
// acc += Ah*Bh + Ah*Bl + Al*Bh  (f32 accumulation; ~fp32 accuracy)

template<int NT, int K>
__global__ void gemm_wmma(const float* __restrict__ A, int lda,
                          const _Float16* __restrict__ BTh,
                          const _Float16* __restrict__ BTl,
                          float* __restrict__ C, int ldc,
                          const float* __restrict__ bias,
                          int M, int accumulate)
{
    const int lane = threadIdx.x & 31;
    const int wid  = threadIdx.x >> 5;
    const int mt   = blockIdx.x * (blockDim.x >> 5) + wid;
    const int m0   = mt * 16;
    if (m0 >= M) return;

    const int mrow   = m0 + (lane & 15);
    const int arowc  = mrow < M ? mrow : (M - 1);
    const int khalfA = (lane & 16) >> 1;   // 0 or 8   (A 16x32 f16 layout K split)
    const int kbaseB = (lane & 16) >> 1;   // 0 or 8   (B 32x16 f16 layout K split)
    const int ncol   = lane & 15;

    // per-lane base pointers; all per-(t,kb) deltas are compile-time immediates
    const float*    abase = A + (size_t)arowc * lda + khalfA;
    const _Float16* bbh   = BTh + (size_t)ncol * K + kbaseB;
    const _Float16* bbl   = BTl + (size_t)ncol * K + kbaseB;

    v8f acc[NT] = {};

#pragma unroll
    for (int kb = 0; kb < K; kb += 32) {
        // ---- A fragment: elems 0..7 -> K=kb+khalf+e, elems 8..15 -> K=kb+16+khalf+e
        float a_raw[16];
        {
            const float4* p = (const float4*)(abase + kb);
            *(float4*)&a_raw[0]  = p[0];
            *(float4*)&a_raw[4]  = p[1];
            *(float4*)&a_raw[8]  = p[4];   // +16 floats
            *(float4*)&a_raw[12] = p[5];
        }
        v16h ah, al;
#pragma unroll
        for (int e = 0; e < 16; ++e) {
            float v = a_raw[e];
            _Float16 h = (_Float16)v;
            ah[e] = h;
            al[e] = (_Float16)(v - (float)h);
        }

#pragma unroll
        for (int t = 0; t < NT; ++t) {
            // ---- B fragment: elems 0..7 -> K=kb+kbaseB+e, elems 8..15 -> K=kb+16+kbaseB+e
            const v8h* ph = (const v8h*)(bbh + t * 16 * K + kb);
            const v8h* pl = (const v8h*)(bbl + t * 16 * K + kb);
            v8h bh0 = ph[0], bh1 = ph[2];
            v8h bl0 = pl[0], bl1 = pl[2];
            v16h bh = __builtin_shufflevector(bh0, bh1, 0,1,2,3,4,5,6,7,8,9,10,11,12,13,14,15);
            v16h bl = __builtin_shufflevector(bl0, bl1, 0,1,2,3,4,5,6,7,8,9,10,11,12,13,14,15);

            acc[t] = __builtin_amdgcn_wmma_f32_16x16x32_f16(false, ah, false, bh,
                                                            (short)0, acc[t], false, false);
            acc[t] = __builtin_amdgcn_wmma_f32_16x16x32_f16(false, ah, false, bl,
                                                            (short)0, acc[t], false, false);
            acc[t] = __builtin_amdgcn_wmma_f32_16x16x32_f16(false, al, false, bh,
                                                            (short)0, acc[t], false, false);
        }
    }

    // ---- C/D layout: VGPR r -> row m0 + r + 8*(lane>=16), col = t*16 + (lane&15)
    const int mbase = m0 + ((lane >> 4) << 3);
#pragma unroll
    for (int r = 0; r < 8; ++r) {
        int m = mbase + r;
        if (m < M) {
            float* crow = C + (size_t)m * ldc + ncol;
#pragma unroll
            for (int t = 0; t < NT; ++t) {
                float v = acc[t][r];
                if (bias) v += bias[t * 16 + ncol];
                if (accumulate) v += crow[t * 16];
                crow[t * 16] = v;
            }
        }
    }
}

// ------------------------------------------------- edge scatter: agg[dst] += h[src]*coef
// One wave per edge; lanes stride channels -> coalesced 128B segments.

__global__ void scatter_kernel(const float* __restrict__ h,
                               const int* __restrict__ src,
                               const int* __restrict__ dst,
                               const float* __restrict__ dinv,
                               float* __restrict__ agg, int E, int H)
{
    int w    = (blockIdx.x * blockDim.x + threadIdx.x) >> 5;
    int lane = threadIdx.x & 31;
    int nw   = (gridDim.x * blockDim.x) >> 5;
    for (int e = w; e < E; e += nw) {
        int s = src[e], d = dst[e];
        float coef = dinv[s] * dinv[d];
        const float* hs = h + (size_t)s * H;
        float* ad = agg + (size_t)d * H;
        for (int c = lane; c < H; c += 32)
            atomicAdd(&ad[c], hs[c] * coef);
    }
}

// ------------------------------------------------- self-loop + bias + LayerNorm + ReLU (+resid)
// One wave per node (H == 128 -> 4 channels/lane), wave32 shuffle reductions.

__global__ void node_post_kernel(const float* __restrict__ agg,
                                 const float* __restrict__ h,
                                 const float* __restrict__ dinv,
                                 const float* __restrict__ bconv,
                                 const float* __restrict__ ln_g,
                                 const float* __restrict__ ln_b,
                                 const float* __restrict__ resid,   // nullable
                                 float* __restrict__ xout,
                                 int M, int H)
{
    int w    = (blockIdx.x * blockDim.x + threadIdx.x) >> 5;
    int lane = threadIdx.x & 31;
    int nw   = (gridDim.x * blockDim.x) >> 5;
    const float invH = 1.0f / (float)H;

    for (int node = w; node < M; node += nw) {
        float di = dinv[node];
        float selfc = 2.0f * di * di;
        const float* ar = agg + (size_t)node * H;
        const float* hr = h   + (size_t)node * H;

        float v[4];
        float sum = 0.0f;
#pragma unroll
        for (int j = 0; j < 4; ++j) {
            int c = lane + j * 32;
            v[j] = ar[c] + hr[c] * selfc + bconv[c];
            sum += v[j];
        }
        for (int off = 16; off > 0; off >>= 1) sum += __shfl_xor(sum, off, 32);
        float mu = sum * invH;

        float var = 0.0f;
#pragma unroll
        for (int j = 0; j < 4; ++j) { float d = v[j] - mu; var += d * d; }
        for (int off = 16; off > 0; off >>= 1) var += __shfl_xor(var, off, 32);
        var *= invH;
        float rs = rsqrtf(var + 1e-5f);

        float* xo = xout + (size_t)node * H;
#pragma unroll
        for (int j = 0; j < 4; ++j) {
            int c = lane + j * 32;
            float y = (v[j] - mu) * rs * ln_g[c] + ln_b[c];
            y = fmaxf(y, 0.0f);
            if (resid) y += resid[(size_t)node * H + c];
            xo[c] = y;
        }
    }
}

// ---------------------------------------------------------------- driver

extern "C" void kernel_launch(void* const* d_in, const int* in_sizes, int n_in,
                              void* d_out, int out_size, void* d_ws, size_t ws_size,
                              hipStream_t stream)
{
    (void)n_in; (void)out_size; (void)ws_size;

    const float* x_in = (const float*)d_in[0];
    const float* WL   = (const float*)d_in[1];
    const float* bL   = (const float*)d_in[2];
    const float* WG   = (const float*)d_in[3];
    const float* bG   = (const float*)d_in[4];
    const float* linW = (const float*)d_in[5];
    const float* linb = (const float*)d_in[6];
    const float* ln_g = (const float*)d_in[7];
    const float* ln_b = (const float*)d_in[8];
    const float* finW = (const float*)d_in[9];
    const float* finb = (const float*)d_in[10];
    const int*   Gei  = (const int*)d_in[11];
    const int*   Lei  = (const int*)d_in[12];

    const int H     = in_sizes[7];           // 128 (compile-time 128 assumed for GEMM K)
    const int OUTC  = in_sizes[10];          // 64
    const int N     = in_sizes[0] / H;       // 50000
    const int DEPTH = in_sizes[2] / H;       // 3
    const int EG    = in_sizes[11] / 2;      // 600000
    const int EL    = in_sizes[12] / 2;      // 400000
    const size_t NH = (size_t)N * H;
    const size_t HH = (size_t)H * H;

    // ---- workspace layout: fp32 buffers, then f16 weight arena
    float* dinvG = (float*)d_ws;
    float* dinvL = dinvG + N;
    float* xcur  = dinvL + N;
    float* hbuf  = xcur + NH;
    float* aggb  = hbuf + NH;
    float* xG    = aggb + NH;
    float* xL    = xG + NH;
    _Float16* wcur = (_Float16*)(xL + NH);

    const int MAXD = 8;
    _Float16 *cWGh[MAXD], *cWGl[MAXD], *cWLh[MAXD], *cWLl[MAXD];
    _Float16 *cL1h[MAXD], *cL1l[MAXD], *cL2h[MAXD], *cL2l[MAXD];
    _Float16 *cFh, *cFl;

    for (int i = 0; i < DEPTH; ++i) {
        cWGh[i] = wcur; wcur += HH;  cWGl[i] = wcur; wcur += HH;
        cWLh[i] = wcur; wcur += HH;  cWLl[i] = wcur; wcur += HH;
        cL1h[i] = wcur; wcur += HH;  cL1l[i] = wcur; wcur += HH;
        cL2h[i] = wcur; wcur += HH;  cL2l[i] = wcur; wcur += HH;
    }
    cFh = wcur; wcur += (size_t)H * OUTC;
    cFl = wcur; wcur += (size_t)H * OUTC;

    // ---- pre-split all weights into transposed f16 hi/lo (tiny: < 1 MB total)
    for (int i = 0; i < DEPTH; ++i) {
        convert_wt_kernel<<<64, 256, 0, stream>>>(WG + i * HH, H, H, H, cWGh[i], cWGl[i]);
        convert_wt_kernel<<<64, 256, 0, stream>>>(WL + i * HH, H, H, H, cWLh[i], cWLl[i]);
        convert_wt_kernel<<<64, 256, 0, stream>>>(linW + (size_t)i * 2 * HH,      H, H, H, cL1h[i], cL1l[i]);
        convert_wt_kernel<<<64, 256, 0, stream>>>(linW + (size_t)i * 2 * HH + HH, H, H, H, cL2h[i], cL2l[i]);
    }
    convert_wt_kernel<<<64, 256, 0, stream>>>(finW, OUTC, H, OUTC, cFh, cFl);

    // ---- symmetric-norm degrees (deg = count(dst) + 2), dinv = deg^-1/2
    fill_kernel<<<256, 256, 0, stream>>>(dinvG, 2.0f, N);
    fill_kernel<<<256, 256, 0, stream>>>(dinvL, 2.0f, N);
    deg_count_kernel<<<(EG + 255) / 256, 256, 0, stream>>>(Gei + EG, dinvG, EG);
    deg_count_kernel<<<(EL + 255) / 256, 256, 0, stream>>>(Lei + EL, dinvL, EL);
    rsqrt_kernel<<<(N + 255) / 256, 256, 0, stream>>>(dinvG, N);
    rsqrt_kernel<<<(N + 255) / 256, 256, 0, stream>>>(dinvL, N);

    hipMemcpyAsync(xcur, x_in, NH * sizeof(float), hipMemcpyDeviceToDevice, stream);

    const int mtiles      = (N + 15) / 16;
    const int gemm_blocks = (mtiles + 3) / 4;   // 4 waves (16-row strips) per 128-thread block

    for (int i = 0; i < DEPTH; ++i) {
        for (int g = 0; g < 2; ++g) {
            const _Float16* Wh  = (g == 0 ? cWGh[i] : cWLh[i]);
            const _Float16* Wl  = (g == 0 ? cWGl[i] : cWLl[i]);
            const float*    bc  = (g == 0 ? bG : bL) + (size_t)i * H;
            const int*      src = (g == 0 ? Gei : Lei);
            const int       E   = (g == 0 ? EG : EL);
            const int*      dst = src + E;
            const float*    dinv = (g == 0 ? dinvG : dinvL);
            float*          xout = (g == 0 ? xG : xL);

            gemm_wmma<8, 128><<<gemm_blocks, 128, 0, stream>>>(xcur, H, Wh, Wl,
                                                               hbuf, H, nullptr, N, 0);
            fill_kernel<<<2048, 256, 0, stream>>>(aggb, 0.0f, (int)NH);
            scatter_kernel<<<2048, 256, 0, stream>>>(hbuf, src, dst, dinv, aggb, E, H);
            node_post_kernel<<<1024, 256, 0, stream>>>(aggb, hbuf, dinv, bc, ln_g, ln_b,
                                                       (i > 0 ? xcur : nullptr), xout, N, H);
        }
        // x = concat([xG, xL]) @ linW[i] + linb[i]   (two GEMMs, second accumulates)
        gemm_wmma<8, 128><<<gemm_blocks, 128, 0, stream>>>(xG, H, cL1h[i], cL1l[i],
                                                           xcur, H, linb + (size_t)i * H, N, 0);
        gemm_wmma<8, 128><<<gemm_blocks, 128, 0, stream>>>(xL, H, cL2h[i], cL2l[i],
                                                           xcur, H, nullptr, N, 1);
    }

    // outputs: (x @ finW + finb, x)
    float* out0 = (float*)d_out;
    float* out1 = out0 + (size_t)N * OUTC;
    gemm_wmma<4, 128><<<gemm_blocks, 128, 0, stream>>>(xcur, H, cFh, cFl,
                                                       out0, OUTC, finb, N, 0);
    hipMemcpyAsync(out1, xcur, NH * sizeof(float), hipMemcpyDeviceToDevice, stream);
}
